// MultiHeadAttention_32667521253714
// MI455X (gfx1250) — matrix operand order repaired and checked by reading, not compiled, not run on hardware
//
#include <hip/hip_runtime.h>
#include <hip/hip_bf16.h>

typedef _Float16 f16;
typedef __attribute__((ext_vector_type(16))) _Float16 v16h;
typedef __attribute__((ext_vector_type(8)))  _Float16 v8h;
typedef __attribute__((ext_vector_type(4)))  _Float16 v4h;
typedef __attribute__((ext_vector_type(8)))  float    v8f;
typedef __attribute__((ext_vector_type(4)))  float    v4f;
typedef __attribute__((ext_vector_type(4)))  unsigned int u32x4;
typedef __attribute__((ext_vector_type(8)))  int          i32x8;
typedef __attribute__((ext_vector_type(4)))  int          i32x4;

#define WMMA_F32_F16(a, b, c) \
  __builtin_amdgcn_wmma_f32_16x16x32_f16(false, (a), false, (b), (short)0, (c), false, false)

#if __has_builtin(__builtin_amdgcn_tensor_load_to_lds) && \
    __has_builtin(__builtin_amdgcn_s_wait_tensorcnt)
#define HAVE_TDM 1
#else
#define HAVE_TDM 0
#endif

// B=4, S=2048, D=1024, H=16, E=64
static constexpr int S = 2048, D = 1024, H = 16, E = 64;

// --- operand builders (contiguous-k layouts; each = two 16B loads) -----------
__device__ __forceinline__ v16h catv(v8h lo, v8h hi) {
  return __builtin_shufflevector(lo, hi, 0, 1, 2, 3, 4, 5, 6, 7,
                                 8, 9, 10, 11, 12, 13, 14, 15);
}
// A (16x32, MxK): lane holds halves K = 16*hi .. 16*hi+15 (contiguous)
__device__ __forceinline__ v16h loadA(const f16* base, int hi) {
  return catv(*(const v8h*)(base + 16 * hi), *(const v8h*)(base + 16 * hi + 8));
}
// B (32x16, KxN): lane holds halves K = 16*hi .. 16*hi+15 (contiguous)
__device__ __forceinline__ v16h loadB(const f16* base, int hi) {
  return catv(*(const v8h*)(base + 16 * hi), *(const v8h*)(base + 16 * hi + 8));
}

#if HAVE_TDM
__device__ __forceinline__ unsigned lds_off(const void* p) {
  return (unsigned)(size_t)p;  // generic LDS pointer: addr[31:0] is the LDS offset
}
// 2D tile load via Tensor Data Mover. tensor_dim == tile_dim (never clips).
// dim0 = contiguous run (elements), dim1 rows with stride0 (elements). f16 data.
__device__ __forceinline__ void tdm_load_2d(unsigned ldsaddr, const f16* gptr,
                                            unsigned d0, unsigned d1,
                                            unsigned long long stride0) {
  unsigned long long ga = (unsigned long long)(size_t)gptr;
  u32x4 g0;
  g0[0] = 1u;                                   // count=1, user descriptor
  g0[1] = ldsaddr;                              // lds_addr
  g0[2] = (unsigned)ga;                         // global_addr[31:0]
  g0[3] = (unsigned)((ga >> 32) & 0x1ffffffull) | (2u << 30);  // addr[56:32], type=2
  i32x8 g1;
  g1[0] = (int)(1u << 16);                                  // data_size=1 (2B)
  g1[1] = (int)((d0 & 0xffffu) << 16);                      // tensor_dim0[15:0]
  g1[2] = (int)((d0 >> 16) | ((d1 & 0xffffu) << 16));       // dim0[31:16], dim1[15:0]
  g1[3] = (int)((d1 >> 16) | (d0 << 16));                   // dim1[31:16], tile_dim0
  g1[4] = (int)(d1 & 0xffffu);                              // tile_dim1 (tile_dim2=0)
  g1[5] = (int)(stride0 & 0xffffffffull);                   // dim0_stride[31:0]
  g1[6] = (int)((stride0 >> 32) & 0xffffull);               // dim0_stride[47:32]
  g1[7] = 0;
  i32x4 z4 = {0, 0, 0, 0};
#if defined(__clang_major__) && __clang_major__ >= 23
  i32x8 z8 = {0, 0, 0, 0, 0, 0, 0, 0};
  __builtin_amdgcn_tensor_load_to_lds(g0, g1, z4, z4, z8, 0);
#else
  __builtin_amdgcn_tensor_load_to_lds(g0, g1, z4, z4, 0);
#endif
}
#endif  // HAVE_TDM

// ---------------------------------------------------------------------------
// Kernel 1: per-head Q/K/V projection.  grid=(S/128, B*H), block=256 (8 waves)
// q,k stored [bh,s,e] f16 ; v stored transposed [bh,e,s] f16
// ---------------------------------------------------------------------------
__global__ __launch_bounds__(256) void mha_qkv_kernel(
    const float* __restrict__ x, const float* __restrict__ Wq,
    const float* __restrict__ Wk, const float* __restrict__ Wv,
    f16* __restrict__ q, f16* __restrict__ k, f16* __restrict__ vt) {
  __shared__ alignas(16) f16 ldsX[128 * 32];     // [row][k]
  __shared__ alignas(16) f16 ldsW[3 * 64 * 32];  // [mat][e][k] (transposed)

  const int tid = threadIdx.x;
  const int w = tid >> 5, l = tid & 31;
  const int st = blockIdx.x;
  const int bh = blockIdx.y;
  const int b = bh >> 4, h = bh & 15;
  const size_t xbase = (size_t)b * S * D + (size_t)st * 128 * D;
  const float* const Wm[3] = {Wq + (size_t)h * D * E, Wk + (size_t)h * D * E,
                              Wv + (size_t)h * D * E};
  const int n_col = l & 15, hi = l >> 4, hi8 = hi * 8;

  v8f acc[3][4] = {};

  for (int d0 = 0; d0 < D; d0 += 32) {
    // stage X tile 128x32: float4 loads -> v4h stores
#pragma unroll
    for (int i = 0; i < 4; ++i) {
      int idx4 = i * 256 + tid;                 // 1024 float4s
      int r = idx4 >> 3, c = (idx4 & 7) * 4;
      v4f t = *(const v4f*)(x + xbase + (size_t)r * D + d0 + c);
      v4h hv = {(f16)t.x, (f16)t.y, (f16)t.z, (f16)t.w};
      *(v4h*)(ldsX + idx4 * 4) = hv;
    }
    // stage W chunks 32x64 transposed -> [e][k]
#pragma unroll
    for (int i = 0; i < 6; ++i) {
      int idx4 = i * 256 + tid;                 // 1536 float4s
      int mat = idx4 >> 9;
      int rem = idx4 & 511;
      int kk = rem >> 4, e4 = (rem & 15) * 4;
      v4f t = *(const v4f*)(Wm[mat] + (size_t)(d0 + kk) * E + e4);
      f16* dst = ldsW + mat * 2048 + e4 * 32 + kk;
      dst[0] = (f16)t.x; dst[32] = (f16)t.y; dst[64] = (f16)t.z; dst[96] = (f16)t.w;
    }
    __syncthreads();

    v16h a = loadA(ldsX + (w * 16 + n_col) * 32, hi);
#pragma unroll
    for (int mat = 0; mat < 3; ++mat) {
#pragma unroll
      for (int nt = 0; nt < 4; ++nt) {
        int e = nt * 16 + n_col;
        v16h bm = loadB(ldsW + mat * 2048 + e * 32, hi);
        acc[mat][nt] = WMMA_F32_F16(a, bm, acc[mat][nt]);
      }
    }
    __syncthreads();
  }

#pragma unroll
  for (int nt = 0; nt < 4; ++nt) {
#pragma unroll
    for (int r = 0; r < 8; ++r) {
      int s = st * 128 + w * 16 + r + hi8;
      int e = nt * 16 + n_col;
      size_t qi = ((size_t)bh * S + s) * E + e;
      q[qi] = (f16)acc[0][nt][r];
      k[qi] = (f16)acc[1][nt][r];
      vt[((size_t)bh * E + e) * S + s] = (f16)acc[2][nt][r];
    }
  }
}

// ---------------------------------------------------------------------------
// Kernel 2: causal flash attention.  grid=(S/32, B*H), block=64 (2 waves)
// S^T = K*Q^T, online softmax, O^T += V^T*P.  K/V tiles streamed by TDM with
// double buffering (TENSORcnt pipeline); LDS reads are ds_load_b128.
// ---------------------------------------------------------------------------
__global__ __launch_bounds__(64) void mha_attn_kernel(
    const f16* __restrict__ q, const f16* __restrict__ k,
    const f16* __restrict__ vt, f16* __restrict__ o) {
#if HAVE_TDM
  __shared__ alignas(16) f16 kvbuf[2][2][4096];  // [wave][buf][K(2048)|V(2048)]
#endif
  __shared__ alignas(16) f16 plds[2 * 512];      // per-wave P scratch [query][key]

  const int tid = threadIdx.x;
  const int w = tid >> 5, l = tid & 31;
  const int qb = blockIdx.x, bh = blockIdx.y;
  const int b = bh >> 4, h = bh & 15;
  const int q_base = qb * 32 + w * 16;
  const int n_col = l & 15, hi = l >> 4, hi8 = hi * 8;
  const int query = q_base + n_col;

  // Q^T as B-matrix (loaded once; e chunks 0..31, 32..63)
  const f16* qrow = q + ((size_t)bh * S + query) * E;
  v16h bq0 = loadB(qrow, hi);
  v16h bq1 = loadB(qrow + 32, hi);

  v8f acc[4] = {};
  float m_run = -1e30f, d_run = 0.f;
  f16* pl = plds + w * 512;
  const int nkb = (q_base + 16 + 31) >> 5;  // causal: keys up to q_base+15

#if HAVE_TDM
  const f16* kg = k + (size_t)bh * S * E;   // [s][e]
  const f16* vg = vt + (size_t)bh * E * S;  // [e][s]
  // preload block 0: K tile is one contiguous 4KB run; V tile is 64 rows x 32
  tdm_load_2d(lds_off(kvbuf[w][0]), kg, 2048u, 1u, 2048ull);
  tdm_load_2d(lds_off(kvbuf[w][0] + 2048), vg, 32u, 64u, (unsigned long long)S);
#endif

  for (int kb = 0; kb < nkb; ++kb) {
    const int k0 = kb * 32;
#if HAVE_TDM
    const f16* cur = kvbuf[w][kb & 1];
    if (kb + 1 < nkb) {
      f16* nxt = kvbuf[w][(kb + 1) & 1];
      const int kn = (kb + 1) * 32;
      // TDM writes are unordered w.r.t. this wave's DS reads: make sure all
      // outstanding LDS reads (from the buffer being recycled) are retired
      // before the DMA starts overwriting it.
      asm volatile("s_wait_dscnt 0x0" ::: "memory");
      tdm_load_2d(lds_off(nxt), kg + (size_t)kn * E, 2048u, 1u, 2048ull);
      tdm_load_2d(lds_off(nxt + 2048), vg + kn, 32u, 64u, (unsigned long long)S);
      __builtin_amdgcn_s_wait_tensorcnt(2);  // next 2 in flight; current done
    } else {
      __builtin_amdgcn_s_wait_tensorcnt(0);
    }
    asm volatile("" ::: "memory");
#endif
    // scores^T: two 16(key)x16(query) tiles, K-dim = E = 64
    v8f sct[2];
#pragma unroll
    for (int t = 0; t < 2; ++t) {
#if HAVE_TDM
      const f16* krow = cur + (t * 16 + n_col) * 64;
#else
      const f16* krow = k + ((size_t)bh * S + k0 + t * 16 + n_col) * E;
#endif
      v16h ak0 = loadA(krow, hi);
      v16h ak1 = loadA(krow + 32, hi);
      v8f z = {};
      z = WMMA_F32_F16(ak0, bq0, z);
      z = WMMA_F32_F16(ak1, bq1, z);
      sct[t] = z;
    }
    // scale + causal mask + online softmax stats
    float sc[16];
    float mloc = -1e30f;
#pragma unroll
    for (int t = 0; t < 2; ++t) {
#pragma unroll
      for (int r = 0; r < 8; ++r) {
        int key = k0 + t * 16 + r + hi8;
        float v = (float)sct[t][r] * 0.125f;  // E^-0.5
        if (key > query) v = -1e30f;
        sc[t * 8 + r] = v;
        mloc = fmaxf(mloc, v);
      }
    }
    mloc = fmaxf(mloc, __shfl_xor(mloc, 16, 32));
    float mnew = fmaxf(m_run, mloc);
    float fsc = __expf(m_run - mnew);
    float rs = 0.f;
#pragma unroll
    for (int t = 0; t < 2; ++t) {
#pragma unroll
      for (int r = 0; r < 8; ++r) {
        float ev = __expf(sc[t * 8 + r] - mnew);
        rs += ev;
        pl[n_col * 32 + t * 16 + r + hi8] = (f16)ev;  // [query][key]
      }
    }
    rs += __shfl_xor(rs, 16, 32);
    d_run = d_run * fsc + rs;
    m_run = mnew;
#pragma unroll
    for (int et = 0; et < 4; ++et) {
#pragma unroll
      for (int r = 0; r < 8; ++r) acc[et][r] *= fsc;
    }
    // P in B layout (32key x 16query) back out of LDS
    v16h bp_ = loadB(pl + n_col * 32, hi);
    // O^T += V^T(16e x 32key) x P
#pragma unroll
    for (int et = 0; et < 4; ++et) {
#if HAVE_TDM
      const f16* vrow = cur + 2048 + (et * 16 + n_col) * 32;
#else
      const f16* vrow = vt + ((size_t)bh * E + et * 16 + n_col) * S + k0;
#endif
      v16h av = loadA(vrow, hi);
      acc[et] = WMMA_F32_F16(av, bp_, acc[et]);
    }
  }

  const float inv = 1.f / d_run;
#pragma unroll
  for (int et = 0; et < 4; ++et) {
#pragma unroll
    for (int r = 0; r < 8; ++r) {
      int e = h * E + et * 16 + r + hi8;
      o[((size_t)b * S + q_base + n_col) * D + e] = (f16)(acc[et][r] * inv);
    }
  }
}

// ---------------------------------------------------------------------------
// Kernel 3: output projection Y = A @ Wp^T + bp.  grid=(BS/128, D/64), block=256
// ---------------------------------------------------------------------------
__global__ __launch_bounds__(256) void mha_proj_kernel(
    const f16* __restrict__ a_in, const float* __restrict__ Wp,
    const float* __restrict__ bp, float* __restrict__ out) {
  const int tid = threadIdx.x;
  const int w = tid >> 5, l = tid & 31;
  const int mb = blockIdx.x, nb = blockIdx.y;
  const int n_col = l & 15, hi = l >> 4, hi8 = hi * 8;
  const int arow = mb * 128 + w * 16 + n_col;
  const int n_base = nb * 64;

  v8f acc[4] = {};
  for (int j0 = 0; j0 < D; j0 += 32) {
    if (j0 + 32 < D) {
      __builtin_prefetch(a_in + (size_t)arow * D + j0 + 32, 0, 0);
      __builtin_prefetch(Wp + (size_t)(n_base + n_col) * D + j0 + 32, 0, 0);
    }
    v16h a = loadA(a_in + (size_t)arow * D + j0, hi);
#pragma unroll
    for (int nt = 0; nt < 4; ++nt) {
      int i = n_base + nt * 16 + n_col;
      const float* wr = Wp + (size_t)i * D + j0 + 16 * hi;  // halves K=16hi..16hi+15
      v4f f0 = *(const v4f*)(wr + 0);
      v4f f1 = *(const v4f*)(wr + 4);
      v4f f2 = *(const v4f*)(wr + 8);
      v4f f3 = *(const v4f*)(wr + 12);
      v16h bm = {(f16)f0.x, (f16)f0.y, (f16)f0.z, (f16)f0.w,
                 (f16)f1.x, (f16)f1.y, (f16)f1.z, (f16)f1.w,
                 (f16)f2.x, (f16)f2.y, (f16)f2.z, (f16)f2.w,
                 (f16)f3.x, (f16)f3.y, (f16)f3.z, (f16)f3.w};
      acc[nt] = WMMA_F32_F16(a, bm, acc[nt]);
    }
  }
#pragma unroll
  for (int nt = 0; nt < 4; ++nt) {
#pragma unroll
    for (int r = 0; r < 8; ++r) {
      int row = mb * 128 + w * 16 + r + hi8;
      int i = n_base + nt * 16 + n_col;
      out[(size_t)row * D + i] = acc[nt][r] + bp[i];
    }
  }
}

// ---------------------------------------------------------------------------
extern "C" void kernel_launch(void* const* d_in, const int* in_sizes, int n_in,
                              void* d_out, int out_size, void* d_ws, size_t ws_size,
                              hipStream_t stream) {
  const float* x  = (const float*)d_in[0];
  const float* Wq = (const float*)d_in[1];
  const float* Wk = (const float*)d_in[2];
  const float* Wv = (const float*)d_in[3];
  const float* Wp = (const float*)d_in[4];
  const float* bp = (const float*)d_in[5];
  float* out = (float*)d_out;

  constexpr size_t QKV_ELEMS = (size_t)4 * H * S * E;  // 8,388,608 per buffer
  f16* qf   = (f16*)d_ws;
  f16* kf   = qf + QKV_ELEMS;
  f16* vtf  = kf + QKV_ELEMS;
  f16* attn = vtf + QKV_ELEMS;  // [B, S, D] f16

  mha_qkv_kernel<<<dim3(S / 128, 4 * H), 256, 0, stream>>>(x, Wq, Wk, Wv, qf, kf, vtf);
  mha_attn_kernel<<<dim3(S / 32, 4 * H), 64, 0, stream>>>(qf, kf, vtf, attn);
  mha_proj_kernel<<<dim3(4 * S / 128, D / 64), 256, 0, stream>>>(attn, Wp, bp, out);
}
